// BasisModel_38912403702035
// MI455X (gfx1250) — matrix operand order repaired and checked by reading, not compiled, not run on hardware
//
#include <hip/hip_runtime.h>

typedef __attribute__((ext_vector_type(16))) _Float16 v16h;
typedef __attribute__((ext_vector_type(8)))  float    v8f;
typedef __attribute__((ext_vector_type(4)))  float    v4f;

#define NB      32      // batches
#define NDIMS   8
#define NDEG    32
#define NPOS    65536
#define WPB     8       // waves per block (256 threads)
#define BPB     64      // blocks per batch

// result = exp(-(x-c_g)^2 / (2*0.04)) @ W^T  ;  -1/(2*0.04) = -12.5
__global__ __launch_bounds__(256) void basis_wmma_kernel(
    const float* __restrict__ weights,    // (B, DIMS, DEGREE) f32
    const float* __restrict__ positions,  // (B, N) f32
    float* __restrict__ out)              // result | zeros, each (B, N, DIMS) f32
{
    const int tid    = threadIdx.x;
    const int lane   = tid & 31;
    const int waveId = tid >> 5;

    const int b     = blockIdx.x / BPB;
    const int chunk = blockIdx.x % BPB;
    const int wavesPerBatch = BPB * WPB;                 // 512
    const int waveInBatch   = chunk * WPB + waveId;

    const int half = lane >> 4;     // 0: lanes 0-15, 1: lanes 16-31
    const int m    = lane & 15;     // A-matrix row (position within tile) == B-matrix column (dim)

    // ---------------- B matrix: W^T, 32x16 f16, contiguous-k layout ----------------
    // lane (half,m): element j holds B[k = half*16 + j][n = m] = weights[b][m][half*16+j]
    v16h bw;
    if (m < NDIMS) {
        const v4f* wrow = (const v4f*)(weights + ((size_t)b * NDIMS + m) * NDEG + half * 16);
        v4f w0 = wrow[0], w1 = wrow[1], w2 = wrow[2], w3 = wrow[3];
        #pragma unroll
        for (int j = 0; j < 4; ++j) {
            bw[j]      = (_Float16)w0[j];
            bw[j + 4]  = (_Float16)w1[j];
            bw[j + 8]  = (_Float16)w2[j];
            bw[j + 12] = (_Float16)w3[j];
        }
    } else {
        #pragma unroll
        for (int j = 0; j < 16; ++j) bw[j] = (_Float16)0.0f;
    }

    // Per-lane RBF centers in A-matrix layout:
    // element j of v16h A maps to K = base + j, base = half*16
    const float step = 1.01f / 31.0f;
    const int   base = half * 16;
    float centers[16];
    #pragma unroll
    for (int j = 0; j < 16; ++j)
        centers[j] = step * (float)(base + j);

    const float*  xsrc  = positions + (size_t)b * NPOS;
    float*        rbase = out + (size_t)b * NPOS * NDIMS;
    float*        zbase = out + (size_t)NB * NPOS * NDIMS + (size_t)b * NPOS * NDIMS;
    const float   rs    = -12.5f;   // -1/(2*SCALE)

    const int nTiles = NPOS / 16;   // 4096 tiles per batch
    for (int t = waveInBatch; t < nTiles; t += wavesPerBatch) {
        const int pos0 = t * 16;

        // warm next tile's positions in cache (gfx1250 global_prefetch_b8)
        if (t + wavesPerBatch < nTiles)
            __builtin_prefetch(xsrc + (t + wavesPerBatch) * 16 + m, 0, 0);

        // -------- A matrix: basis tile 16x32 f16, generated in-register --------
        const float x = xsrc[pos0 + m];
        v16h a;
        #pragma unroll
        for (int j = 0; j < 16; ++j) {
            const float d = x - centers[j];
            a[j] = (_Float16)__expf(rs * d * d);
        }

        // -------- D = A x B (16x16x32 f16 -> f32) --------
        v8f c = {};
        c = __builtin_amdgcn_wmma_f32_16x16x32_f16(
                /*neg_a=*/false, a, /*neg_b=*/false, bw,
                /*c_mod=*/(short)0, c, /*reuse_a=*/false, /*reuse_b=*/false);

        // -------- store result: lane holds column d=m (valid d<8), rows r+8*half --------
        if (m < NDIMS) {
            float* orow = rbase + (size_t)(pos0 + half * 8) * NDIMS + m;
            #pragma unroll
            for (int r = 0; r < 8; ++r)
                orow[(size_t)r * NDIMS] = c[r];
        }

        // -------- zeros output: one coalesced float4 per lane (512B per tile) --------
        v4f z = {0.0f, 0.0f, 0.0f, 0.0f};
        *(v4f*)(zbase + (size_t)pos0 * NDIMS + lane * 4) = z;
    }
}

extern "C" void kernel_launch(void* const* d_in, const int* in_sizes, int n_in,
                              void* d_out, int out_size, void* d_ws, size_t ws_size,
                              hipStream_t stream) {
    (void)in_sizes; (void)n_in; (void)d_ws; (void)ws_size; (void)out_size;
    const float* weights   = (const float*)d_in[0];
    // d_in[1] = weights_std: unused by the reference forward pass
    const float* positions = (const float*)d_in[2];
    float* out = (float*)d_out;

    dim3 grid(NB * BPB);   // 2048 blocks x 256 threads = 16384 waves
    dim3 block(WPB * 32);
    basis_wmma_kernel<<<grid, block, 0, stream>>>(weights, positions, out);
}
